// MORTM_90503550861976
// MI455X (gfx1250) — matrix operand order repaired and checked by reading, not compiled, not run on hardware
//
#include <hip/hip_runtime.h>
#include <math.h>

#define D_MODEL 1024
#define INTER   1024
#define NEXP    8
#define TOKENS  8192
#define MTILE   32
#define MAT_ELEMS (1u << 20)   // 1024*1024 elements per weight matrix
#define N_MATS  27             // w1[8], w3[8], w2[8], sw1, sw3, sw2

typedef float  v8f   __attribute__((ext_vector_type(8)));
typedef __bf16 v16bf __attribute__((ext_vector_type(16)));

__device__ __forceinline__ v8f wmma_bf16(v16bf a, v16bf b, v8f c) {
  // D(16x16,f32) = A(16x32,bf16) * B(32x16,bf16) + C
  return __builtin_amdgcn_wmma_f32_16x16x32_bf16(false, a, false, b, (short)0, c, false, false);
}

__device__ __forceinline__ v16bf ldfrag(const __bf16* p) {
  return *reinterpret_cast<const v16bf*>(p);   // 32B -> 2x b128
}

// A-fragment (16-bit A 16x32, ISA 7.12.2): lane = m + 16*h holds
//   V0..V3 -> K = 8h + 0..7 ; V4..V7 -> K = 8h + 16..23
__device__ __forceinline__ int a_elem_of_kin(int kin) {           // kin 0..31 -> e 0..15
  return (kin < 16) ? (kin & 7) : (8 + (kin & 7));
}
__device__ __forceinline__ int a_half_of_kin(int kin) { return (kin >> 3) & 1; }

// ---------------------------------------------------------------------------
__global__ void init_counts_kernel(int* counts) {
  if (threadIdx.x < NEXP) counts[threadIdx.x] = 0;
}

// ---------------------------------------------------------------------------
// Gate: softmax over 8 experts, top-2, build gathered per-expert lists.
// ---------------------------------------------------------------------------
__global__ void gate_kernel(const float* __restrict__ x,
                            const float* __restrict__ gw,
                            int* __restrict__ counts,
                            int* __restrict__ tok_lists,
                            float* __restrict__ wt_lists) {
  const int warp = threadIdx.x >> 5;
  const int lane = threadIdx.x & 31;
  const int t = blockIdx.x * 8 + warp;
  const int e = lane & 7;
  const int seg = lane >> 3;

  const float* xr = x + (size_t)t * D_MODEL + seg * 256;
  const float* gr = gw + (size_t)e * D_MODEL + seg * 256;
  float s = 0.f;
  #pragma unroll 8
  for (int d = 0; d < 256; ++d) s = fmaf(xr[d], gr[d], s);
  s += __shfl_xor(s, 8);
  s += __shfl_xor(s, 16);

  float m = s;
  m = fmaxf(m, __shfl_xor(m, 1));
  m = fmaxf(m, __shfl_xor(m, 2));
  m = fmaxf(m, __shfl_xor(m, 4));
  float p = __expf(s - m);
  float sum = p;
  sum += __shfl_xor(sum, 1);
  sum += __shfl_xor(sum, 2);
  sum += __shfl_xor(sum, 4);
  const float prob = p / sum;

  float m1 = prob;
  m1 = fmaxf(m1, __shfl_xor(m1, 1));
  m1 = fmaxf(m1, __shfl_xor(m1, 2));
  m1 = fmaxf(m1, __shfl_xor(m1, 4));
  int c1 = (prob == m1) ? e : NEXP;
  c1 = min(c1, __shfl_xor(c1, 1));
  c1 = min(c1, __shfl_xor(c1, 2));
  c1 = min(c1, __shfl_xor(c1, 4));

  const float prob2 = (e == c1) ? -1.0f : prob;
  float m2 = prob2;
  m2 = fmaxf(m2, __shfl_xor(m2, 1));
  m2 = fmaxf(m2, __shfl_xor(m2, 2));
  m2 = fmaxf(m2, __shfl_xor(m2, 4));
  int c2 = (prob2 == m2) ? e : NEXP;
  c2 = min(c2, __shfl_xor(c2, 1));
  c2 = min(c2, __shfl_xor(c2, 2));
  c2 = min(c2, __shfl_xor(c2, 4));

  if (lane == 0) {
    int p1 = atomicAdd(&counts[c1], 1);
    tok_lists[c1 * TOKENS + p1] = t;
    wt_lists[c1 * TOKENS + p1] = m1;
    int p2 = atomicAdd(&counts[c2], 1);
    tok_lists[c2 * TOKENS + p2] = t;
    wt_lists[c2 * TOKENS + p2] = m2;
  }
}

// ---------------------------------------------------------------------------
// Split all weight matrices into hi/lo bf16 planes, pre-swizzled into the
// B-fragment layout: elem index = ((nt*32 + kt)*32 + lane)*16 + e, where
// lane = (n&15) + 16*h and B[k][n] sits at e = (k&7) + 8*((k>>4)&1) with h = (k>>3)&1.
// Each thread produces one VGPR pair (e = 2j, 2j+1 -> rows k0, k0+1).
// ---------------------------------------------------------------------------
__global__ void convert_weights_kernel(const float* __restrict__ w1,
                                       const float* __restrict__ w3,
                                       const float* __restrict__ w2,
                                       const float* __restrict__ sw1,
                                       const float* __restrict__ sw3,
                                       const float* __restrict__ sw2,
                                       __bf16* __restrict__ wh,
                                       __bf16* __restrict__ wl) {
  const int mat = blockIdx.y;
  const float* src;
  if      (mat <  8) src = w1  + (size_t)mat        * MAT_ELEMS;
  else if (mat < 16) src = w3  + (size_t)(mat - 8)  * MAT_ELEMS;
  else if (mat < 24) src = w2  + (size_t)(mat - 16) * MAT_ELEMS;
  else if (mat == 24) src = sw1;
  else if (mat == 25) src = sw3;
  else               src = sw2;

  const int p    = blockIdx.x * 256 + threadIdx.x;  // 0 .. 512K-1 (pairs)
  const int j    = p & 7;                           // VGPR index
  const int lane = (p >> 3) & 31;
  const int nt   = (p >> 8) & 63;
  const int kt   = p >> 14;                         // 0..31
  const int h    = lane >> 4;
  const int k0   = kt * 32 + 8 * h + 2 * j + (j >= 4 ? 8 : 0);  // B layout: K = 8h + e (e<8), 16 + 8h + (e-8) (e>=8)
  const int n    = nt * 16 + (lane & 15);

  const float v0 = src[(size_t)k0 * 1024 + n];
  const float v1 = src[(size_t)(k0 + 1) * 1024 + n];
  const __bf16 h0 = (__bf16)v0; const __bf16 l0 = (__bf16)(v0 - (float)h0);
  const __bf16 h1 = (__bf16)v1; const __bf16 l1 = (__bf16)(v1 - (float)h1);

  const size_t dst = (size_t)mat * MAT_ELEMS +
                     (((size_t)(nt * 32 + kt) * 32 + lane) * 16 + 2 * j);
  wh[dst] = h0; wh[dst + 1] = h1;
  wl[dst] = l0; wl[dst + 1] = l1;
}

// ---------------------------------------------------------------------------
// SwiGLU MLP over a 32-token gathered tile, split-bf16 WMMA (3 wmma / K=32).
// ---------------------------------------------------------------------------
template <bool SHARED>
__global__ void __launch_bounds__(256)
moe_mlp_kernel(const float* __restrict__ x,
               const __bf16* __restrict__ wh,
               const __bf16* __restrict__ wl,
               const float* __restrict__ B1g,
               const float* __restrict__ B2g,
               const float* __restrict__ B3g,
               const int* __restrict__ counts,
               const int* __restrict__ tok_lists,
               const float* __restrict__ wt_lists,
               float* __restrict__ out) {
  extern __shared__ char smem_raw[];
  __bf16* Xh = (__bf16*)smem_raw;       // [mt2][kt32][lane32][e16]
  __bf16* Xl = Xh + 32768;
  __bf16* Hh = Xl + 32768;
  __bf16* Hl = Hh + 32768;
  int*    tokS = (int*)(Hl + 32768);
  float*  wtS  = (float*)(tokS + MTILE);

  const int e     = SHARED ? 0 : blockIdx.y;
  const int count = SHARED ? TOKENS : counts[e];
  const int base  = blockIdx.x * MTILE;
  if (base >= count) return;                       // uniform per block

  const size_t m1 = (size_t)(SHARED ? 24 : e)      * MAT_ELEMS;
  const size_t m3 = (size_t)(SHARED ? 25 : 8 + e)  * MAT_ELEMS;
  const size_t m2 = (size_t)(SHARED ? 26 : 16 + e) * MAT_ELEMS;
  const float* B1 = B1g + (SHARED ? 0 : e * INTER);
  const float* B3 = B3g + (SHARED ? 0 : e * INTER);
  const float* B2 = B2g + (SHARED ? 0 : e * D_MODEL);

  const int tid = threadIdx.x;
  if (tid < MTILE) {
    const int r = base + tid;
    if (SHARED)         { tokS[tid] = r;                         wtS[tid] = 1.0f; }
    else if (r < count) { tokS[tid] = tok_lists[e * TOKENS + r]; wtS[tid] = wt_lists[e * TOKENS + r]; }
    else                { tokS[tid] = 0;                         wtS[tid] = 0.0f; }
  }
  __syncthreads();

  // gather + split X into A-fragment-layout LDS planes
  for (int i = tid; i < MTILE * D_MODEL; i += 256) {
    const int row = i >> 10, k = i & (D_MODEL - 1);
    const float v = x[(size_t)tokS[row] * D_MODEL + k];
    const __bf16 hi = (__bf16)v;
    const __bf16 lo = (__bf16)(v - (float)hi);
    const int mt = row >> 4, kt = k >> 5, kin = k & 31;
    const int a = (((mt * 32 + kt) * 32) + ((row & 15) + 16 * a_half_of_kin(kin))) * 16
                + a_elem_of_kin(kin);
    Xh[a] = hi; Xl[a] = lo;
  }
  __syncthreads();

  const int warp = tid >> 5, lane = tid & 31;
  const int laneN = lane & 15, laneH = lane >> 4;

  // ---------------- GEMM1: H = silu(X*W1 + b1) * (X*W3 + b3) ----------------
  // one N-tile per sub-chunk: 4 accums + 4 A frags + 4 B frags live (~96 VGPRs)
  #pragma unroll 1
  for (int nc = 0; nc < 8; ++nc) {
    const int nt = warp * 8 + nc;
    v8f acc1[2] = {};
    v8f acc3[2] = {};
    #pragma unroll 1
    for (int kt = 0; kt < 32; ++kt) {
      v16bf xh[2], xl[2];
      #pragma unroll
      for (int mt = 0; mt < 2; ++mt) {
        const int ao = ((mt * 32 + kt) * 32 + lane) * 16;
        xh[mt] = ldfrag(Xh + ao);
        xl[mt] = ldfrag(Xl + ao);
      }
      const size_t bo = ((size_t)(nt * 32 + kt) * 32 + lane) * 16;
      const v16bf b1h = ldfrag(wh + m1 + bo);
      const v16bf b1l = ldfrag(wl + m1 + bo);
      const v16bf b3h = ldfrag(wh + m3 + bo);
      const v16bf b3l = ldfrag(wl + m3 + bo);
      #pragma unroll
      for (int mt = 0; mt < 2; ++mt) {
        acc1[mt] = wmma_bf16(xh[mt], b1h, acc1[mt]);
        acc1[mt] = wmma_bf16(xh[mt], b1l, acc1[mt]);
        acc1[mt] = wmma_bf16(xl[mt], b1h, acc1[mt]);
        acc3[mt] = wmma_bf16(xh[mt], b3h, acc3[mt]);
        acc3[mt] = wmma_bf16(xh[mt], b3l, acc3[mt]);
        acc3[mt] = wmma_bf16(xl[mt], b3h, acc3[mt]);
      }
    }
    // epilogue: SwiGLU, split, store H in A-fragment layout
    const int n = nt * 16 + laneN;
    const float bb1 = B1[n], bb3 = B3[n];
    const int kt2 = n >> 5, kin2 = n & 31;
    const int h2 = a_half_of_kin(kin2), e2 = a_elem_of_kin(kin2);
    #pragma unroll
    for (int mt = 0; mt < 2; ++mt) {
      #pragma unroll
      for (int r = 0; r < 8; ++r) {
        const int rloc = r + 8 * laneH;              // 0..15 within tile
        const float g = acc1[mt][r] + bb1;
        const float u = acc3[mt][r] + bb3;
        // silu via 1-ulp v_rcp_f32: error far below the split-bf16 floor
        const float sg = __builtin_amdgcn_rcpf(1.0f + __expf(-g));
        const float hv = g * sg * u;
        const __bf16 hh = (__bf16)hv;
        const __bf16 hl = (__bf16)(hv - (float)hh);
        const int a = (((mt * 32 + kt2) * 32) + (rloc + 16 * h2)) * 16 + e2;
        Hh[a] = hh; Hl[a] = hl;
      }
    }
  }
  __syncthreads();

  // ---------------- GEMM2: Y = H*W2 (+ b2), scatter ----------------
  #pragma unroll 1
  for (int nc = 0; nc < 4; ++nc) {
    const int nt0 = warp * 8 + nc * 2;
    v8f acc[2][2] = {};
    #pragma unroll 1
    for (int kt = 0; kt < 32; ++kt) {
      v16bf ah[2], al[2];
      #pragma unroll
      for (int mt = 0; mt < 2; ++mt) {
        const int ao = ((mt * 32 + kt) * 32 + lane) * 16;
        ah[mt] = ldfrag(Hh + ao);
        al[mt] = ldfrag(Hl + ao);
      }
      #pragma unroll
      for (int nn = 0; nn < 2; ++nn) {
        const size_t bo = ((size_t)((nt0 + nn) * 32 + kt) * 32 + lane) * 16;
        const v16bf b2h = ldfrag(wh + m2 + bo);
        const v16bf b2l = ldfrag(wl + m2 + bo);
        #pragma unroll
        for (int mt = 0; mt < 2; ++mt) {
          acc[mt][nn] = wmma_bf16(ah[mt], b2h, acc[mt][nn]);
          acc[mt][nn] = wmma_bf16(ah[mt], b2l, acc[mt][nn]);
          acc[mt][nn] = wmma_bf16(al[mt], b2h, acc[mt][nn]);
        }
      }
    }
    #pragma unroll
    for (int nn = 0; nn < 2; ++nn) {
      const int n = (nt0 + nn) * 16 + laneN;
      const float bb2 = B2[n];
      #pragma unroll
      for (int mt = 0; mt < 2; ++mt) {
        #pragma unroll
        for (int r = 0; r < 8; ++r) {
          const int row = mt * 16 + r + 8 * laneH;
          const int tok = tokS[row];
          const float y = acc[mt][nn][r] + bb2;
          if (SHARED) {
            out[(size_t)tok * D_MODEL + n] = y;
          } else {
            const float c = wtS[row];
            if (c != 0.0f) {
              __hip_atomic_fetch_add(&out[(size_t)tok * D_MODEL + n], c * y,
                                     __ATOMIC_RELAXED, __HIP_MEMORY_SCOPE_AGENT);
            }
          }
        }
      }
    }
  }
}

// ---------------------------------------------------------------------------
extern "C" void kernel_launch(void* const* d_in, const int* in_sizes, int n_in,
                              void* d_out, int out_size, void* d_ws, size_t ws_size,
                              hipStream_t stream) {
  const float* x      = (const float*)d_in[0];
  const float* gate_w = (const float*)d_in[1];
  const float* w1     = (const float*)d_in[2];
  const float* b1     = (const float*)d_in[3];
  const float* w2     = (const float*)d_in[4];
  const float* b2     = (const float*)d_in[5];
  const float* w3     = (const float*)d_in[6];
  const float* b3     = (const float*)d_in[7];
  const float* sw1    = (const float*)d_in[8];
  const float* sb1    = (const float*)d_in[9];
  const float* sw2    = (const float*)d_in[10];
  const float* sb2    = (const float*)d_in[11];
  const float* sw3    = (const float*)d_in[12];
  const float* sb3    = (const float*)d_in[13];
  float* out = (float*)d_out;

  char* ws = (char*)d_ws;
  int*    counts    = (int*)ws;
  int*    tok_lists = (int*)(ws + 256);
  float*  wt_lists  = (float*)(ws + 256 + (size_t)NEXP * TOKENS * sizeof(int));
  __bf16* wh        = (__bf16*)(ws + (1 << 20));
  __bf16* wl        = wh + (size_t)N_MATS * MAT_ELEMS;

  const size_t smem = 4u * 32768u * sizeof(__bf16) + MTILE * (sizeof(int) + sizeof(float));

  init_counts_kernel<<<1, 64, 0, stream>>>(counts);
  gate_kernel<<<TOKENS / 8, 256, 0, stream>>>(x, gate_w, counts, tok_lists, wt_lists);
  convert_weights_kernel<<<dim3(MAT_ELEMS / 2 / 256, N_MATS), 256, 0, stream>>>(
      w1, w3, w2, sw1, sw3, sw2, wh, wl);
  // shared expert first: plain stores initialize d_out
  moe_mlp_kernel<true><<<dim3(TOKENS / MTILE), 256, smem, stream>>>(
      x, wh, wl, sb1, sb2, sb3, nullptr, nullptr, nullptr, out);
  // routed experts: gathered tiles, weighted atomic accumulation
  moe_mlp_kernel<false><<<dim3(TOKENS / MTILE, NEXP), 256, smem, stream>>>(
      x, wh, wl, b1, b2, b3, counts, tok_lists, wt_lists, out);
}
